// Attention_1606317768853
// MI455X (gfx1250) — compile-verified, matrix operand order checked by reading, not run on hardware
//
#include <hip/hip_runtime.h>
#include <cstdint>
#include <cstddef>

// ---------------------------------------------------------------------------
// Problem constants (from reference)
// ---------------------------------------------------------------------------
constexpr int Bc    = 2;
constexpr int Sc    = 2048;
constexpr int DIMc  = 2048;
constexpr int NHc   = 16;
constexpr int NKVc  = 4;
constexpr int HDc   = 128;           // head dim
constexpr int NREPc = NHc / NKVc;    // 4
constexpr int Mrows = Bc * Sc;       // 4096 rows for all GEMMs
constexpr float SCALEc = 0.08838834764831845f;  // 1/sqrt(128)

typedef __bf16 bf16_t;
typedef __attribute__((ext_vector_type(16))) __bf16 v16bf;
typedef __attribute__((ext_vector_type(8)))  __bf16 v8bf;
typedef __attribute__((ext_vector_type(8)))  float  v8f;

#define DEVFN static __device__ __forceinline__

// ---------------------------------------------------------------------------
// WMMA helpers (CDNA5 16x16x32 bf16, f32 accumulate)
//
// A fragment (16x32, row-major source, stride K):
//   lane<16 : row = lane,    elems 0..7 = K[0..7],  elems 8..15 = K[16..23]
//   lane>=16: row = lane-16, elems 0..7 = K[8..15], elems 8..15 = K[24..31]
// B fragment (32x16) mirrors A with N in place of M, sourced from a
// [N x K] row-major (i.e. pre-transposed) buffer with the same pattern.
// C/D fragment (16x16 f32): lane<16 -> col=lane,   row = m0 + r
//                           lane>=16-> col=lane-16,row = m0 + 8 + r
// ---------------------------------------------------------------------------
DEVFN v16bf load_frag(const bf16_t* p /* row start + k0 */, int lane) {
    const int kOff = (lane & 16) ? 8 : 0;
    v8bf lo = *(const v8bf*)(p + kOff);
    v8bf hi = *(const v8bf*)(p + 16 + kOff);
    return __builtin_shufflevector(lo, hi, 0, 1, 2, 3, 4, 5, 6, 7,
                                           8, 9, 10, 11, 12, 13, 14, 15);
}

DEVFN v8f wmma_bf16(v16bf a, v16bf b, v8f c) {
    return __builtin_amdgcn_wmma_f32_16x16x32_bf16(
        /*neg_a=*/false, a, /*neg_b=*/false, b,
        /*c_mod=*/(short)0, c, /*reuse_a=*/false, /*reuse_b=*/false);
}

DEVFN v8f vzero8() { return (v8f){0.f, 0.f, 0.f, 0.f, 0.f, 0.f, 0.f, 0.f}; }

DEVFN v16bf ones_frag() {
    v16bf v;
#pragma unroll
    for (int i = 0; i < 16; ++i) v[i] = (bf16_t)1.0f;
    return v;
}

// ---------------------------------------------------------------------------
// Elementwise prep kernels
// ---------------------------------------------------------------------------
__global__ __launch_bounds__(256) void cast_f32_bf16(const float* __restrict__ x,
                                                     bf16_t* __restrict__ y,
                                                     size_t n) {
    size_t i = (size_t)blockIdx.x * blockDim.x + threadIdx.x;
    if (i < n) y[i] = (bf16_t)x[i];
}

// w: [K, N] fp32 row-major  ->  wt: [N, K] bf16 row-major
__global__ __launch_bounds__(256) void cast_transpose(const float* __restrict__ w,
                                                      bf16_t* __restrict__ wt,
                                                      int K, int N) {
    size_t i = (size_t)blockIdx.x * blockDim.x + threadIdx.x;
    size_t total = (size_t)K * N;
    if (i >= total) return;
    int k = (int)(i % (size_t)K);
    int n = (int)(i / (size_t)K);
    wt[(size_t)n * K + k] = (bf16_t)w[(size_t)k * N + n];
}

// t0: [B, S, nHeads*HD] bf16 -> tr: [B, nHeads, S, HD] bf16 with RoPE.
// `scale` folds the 1/sqrt(HD) attention scale into Q (1.0 for K).
__global__ __launch_bounds__(256) void rope_rearrange(const bf16_t* __restrict__ t0,
                                                      const float* __restrict__ cosp,
                                                      const float* __restrict__ sinp,
                                                      bf16_t* __restrict__ tr,
                                                      int nHeads, float scale) {
    size_t i = (size_t)blockIdx.x * blockDim.x + threadIdx.x;
    size_t total = (size_t)Bc * Sc * nHeads * (HDc / 2);
    if (i >= total) return;
    int d2 = (int)(i % (HDc / 2));
    size_t t = i / (HDc / 2);
    int hh = (int)(t % nHeads); t /= nHeads;
    int s  = (int)(t % Sc);
    int b  = (int)(t / Sc);
    size_t src = ((size_t)((size_t)b * Sc + s) * nHeads + hh) * HDc + 2 * d2;
    float xr = (float)t0[src];
    float xi = (float)t0[src + 1];
    float c  = cosp[(size_t)s * (HDc / 2) + d2];
    float sn = sinp[(size_t)s * (HDc / 2) + d2];
    size_t dst = (((size_t)b * nHeads + hh) * Sc + s) * (size_t)HDc + 2 * d2;
    tr[dst]     = (bf16_t)((xr * c - xi * sn) * scale);
    tr[dst + 1] = (bf16_t)((xr * sn + xi * c) * scale);
}

// v0: [B, S, KVH*HD] -> vt: [B, KVH, HD, S]  (feature-major for PV B-fragments)
__global__ __launch_bounds__(256) void v_transpose(const bf16_t* __restrict__ v0,
                                                   bf16_t* __restrict__ vt) {
    size_t i = (size_t)blockIdx.x * blockDim.x + threadIdx.x;
    size_t total = (size_t)Bc * Sc * NKVc * HDc;
    if (i >= total) return;
    int d = (int)(i % HDc);
    size_t t = i / HDc;
    int kv = (int)(t % NKVc); t /= NKVc;
    int s  = (int)(t % Sc);
    int b  = (int)(t / Sc);
    vt[(((size_t)b * NKVc + kv) * HDc + d) * (size_t)Sc + s] =
        v0[((size_t)((size_t)b * Sc + s) * NKVc + kv) * HDc + d];
}

// ---------------------------------------------------------------------------
// GEMM: C[M,N] = A[M,K] @ B[K,N], with B supplied pre-transposed as BT[N,K].
// 256 threads = 8 waves arranged 2(M) x 4(N); each wave computes a 32x64
// register tile (2 A-frags x 4 B-frags -> 8 WMMAs per 32-deep k-step, 1.5
// b128 loads per WMMA, 8 independent accumulator chains).
// Workgroup covers a 64x256 output tile.
// ---------------------------------------------------------------------------
template <typename OutT>
__global__ __launch_bounds__(256) void gemm_bf16_k(const bf16_t* __restrict__ A,
                                                   const bf16_t* __restrict__ BT,
                                                   OutT* __restrict__ C,
                                                   int M, int N, int K) {
    const int lane = threadIdx.x & 31;
    const int wave = threadIdx.x >> 5;
    const int wm   = wave & 1;       // 0..1
    const int wn   = wave >> 1;      // 0..3
    const int l16  = lane & 15;
    const int m0   = blockIdx.y * 64 + wm * 32;
    const int n0   = blockIdx.x * 256 + wn * 64;

    const bf16_t* aRow0 = A + (size_t)(m0 + l16) * K;
    const bf16_t* aRow1 = A + (size_t)(m0 + 16 + l16) * K;
    const bf16_t* bRow[4];
#pragma unroll
    for (int t = 0; t < 4; ++t)
        bRow[t] = BT + (size_t)(n0 + t * 16 + l16) * K;

    v8f acc[2][4];
#pragma unroll
    for (int mi = 0; mi < 2; ++mi)
#pragma unroll
        for (int t = 0; t < 4; ++t)
            acc[mi][t] = vzero8();

    for (int k0 = 0; k0 < K; k0 += 32) {
        v16bf a0 = load_frag(aRow0 + k0, lane);
        v16bf a1 = load_frag(aRow1 + k0, lane);
#pragma unroll
        for (int t = 0; t < 4; ++t) {
            v16bf b = load_frag(bRow[t] + k0, lane);
            acc[0][t] = wmma_bf16(a0, b, acc[0][t]);
            acc[1][t] = wmma_bf16(a1, b, acc[1][t]);
        }
    }

    const int rowHalf = (lane & 16) ? 8 : 0;
#pragma unroll
    for (int mi = 0; mi < 2; ++mi) {
#pragma unroll
        for (int t = 0; t < 4; ++t) {
            const int col = n0 + t * 16 + l16;
#pragma unroll
            for (int r = 0; r < 8; ++r) {
                const int row = m0 + mi * 16 + rowHalf + r;
                C[(size_t)row * N + col] = (OutT)acc[mi][t][r];
            }
        }
    }
}

// ---------------------------------------------------------------------------
// Flash attention: one wave per (q-block of 16 rows, head, batch).
// qr: [B, H, S, HD] (pre-scaled)  kr: [B, KVH, S, HD]  vt: [B, KVH, HD, S]
// ao: [B*S, H*HD] bf16 (feeds the output-projection GEMM)
//
// Key loop is split: fully-unmasked steady-state blocks (no cmp/cndmask)
// and a masked causal tail (1 block per q-tile).
// Row-sums of P are computed by an extra WMMA against an all-ones B
// fragment (C-layout row sums land exactly in the l[] register layout),
// eliminating the 32-op ds_bpermute sum reduction per block.
// ---------------------------------------------------------------------------
template <bool MASKED>
DEVFN void attn_kblock(int key0, int qb, int lane, int l16, int half, int kOff,
                       const bf16_t* __restrict__ kBase,
                       const bf16_t* __restrict__ vBase,
                       const v16bf (&qf)[4], const v16bf ones,
                       v8f (&o)[8], float (&m)[8], float (&l)[8],
                       bf16_t* pLds) {
    // scores for keys [key0, key0+16) and [key0+16, key0+32)
    v8f s0 = vzero8();
    v8f s1 = vzero8();
#pragma unroll
    for (int c = 0; c < 4; ++c) {
        v16bf kf = load_frag(kBase + (size_t)(key0 + l16) * HDc + c * 32, lane);
        s0 = wmma_bf16(qf[c], kf, s0);
    }
#pragma unroll
    for (int c = 0; c < 4; ++c) {
        v16bf kf = load_frag(kBase + (size_t)(key0 + 16 + l16) * HDc + c * 32, lane);
        s1 = wmma_bf16(qf[c], kf, s1);
    }

    // online softmax max-update (row-wise over 16-lane groups)
    float alpha[8], p0v[8], p1v[8];
#pragma unroll
    for (int r = 0; r < 8; ++r) {
        float sa = s0[r];
        float sb = s1[r];
        if (MASKED) {
            const int row = qb * 16 + half + r;
            const int kA = key0 + l16;
            if (kA > row)      sa = -1e30f;
            if (kA + 16 > row) sb = -1e30f;
        }
        float mx = fmaxf(sa, sb);
#pragma unroll
        for (int d = 1; d < 16; d <<= 1)
            mx = fmaxf(mx, __shfl_xor(mx, d, 32));
        const float mn = fmaxf(m[r], mx);
        alpha[r] = __expf(m[r] - mn);
        p0v[r] = __expf(sa - mn);
        p1v[r] = __expf(sb - mn);
        m[r] = mn;
    }

    // rescale accumulators
#pragma unroll
    for (int f = 0; f < 8; ++f)
#pragma unroll
        for (int r = 0; r < 8; ++r)
            o[f][r] = o[f][r] * alpha[r];

    // repack P (C-layout) -> A-layout through per-wave LDS tile
#pragma unroll
    for (int r = 0; r < 8; ++r) {
        pLds[(half + r) * 32 + l16]      = (bf16_t)p0v[r];
        pLds[(half + r) * 32 + 16 + l16] = (bf16_t)p1v[r];
    }
    asm volatile("s_wait_dscnt 0" ::: "memory");
    v16bf pf;
    {
        v8bf lo = *(const v8bf*)(pLds + l16 * 32 + kOff);
        v8bf hi = *(const v8bf*)(pLds + l16 * 32 + 16 + kOff);
        pf = __builtin_shufflevector(lo, hi, 0, 1, 2, 3, 4, 5, 6, 7,
                                             8, 9, 10, 11, 12, 13, 14, 15);
    }

    // l update: row-sum of P via WMMA against all-ones B fragment.
    // ls[r] = sum over 32 keys of P[row r], replicated across columns.
    v8f ls = wmma_bf16(pf, ones, vzero8());
#pragma unroll
    for (int r = 0; r < 8; ++r)
        l[r] = l[r] * alpha[r] + ls[r];

    // O += P @ V   (B fragments from feature-major vt, contiguous in key)
#pragma unroll
    for (int f = 0; f < 8; ++f) {
        const bf16_t* vRow = vBase + (size_t)(f * 16 + l16) * Sc + key0;
        v16bf vf = load_frag(vRow, lane);
        o[f] = wmma_bf16(pf, vf, o[f]);
    }
}

__global__ __launch_bounds__(32) void attn_fwd(const bf16_t* __restrict__ qr,
                                               const bf16_t* __restrict__ kr,
                                               const bf16_t* __restrict__ vt,
                                               bf16_t* __restrict__ ao) {
    const int lane = threadIdx.x;
    const int l16  = lane & 15;
    const int half = (lane & 16) ? 8 : 0;
    const int kOff = (lane & 16) ? 8 : 0;
    const int qb = blockIdx.x;          // S/16 q-blocks
    const int h  = blockIdx.y;
    const int b  = blockIdx.z;
    const int kv = h / NREPc;

    __shared__ __attribute__((aligned(16))) bf16_t pLds[16 * 32];

    const bf16_t* qBase = qr + ((size_t)b * NHc  + h ) * Sc * (size_t)HDc;
    const bf16_t* kBase = kr + ((size_t)b * NKVc + kv) * Sc * (size_t)HDc;
    const bf16_t* vBase = vt + ((size_t)b * NKVc + kv) * HDc * (size_t)Sc;

    // Q fragments: 4 chunks of K=32 covering HD=128 (already scaled)
    v16bf qf[4];
#pragma unroll
    for (int c = 0; c < 4; ++c)
        qf[c] = load_frag(qBase + (size_t)(qb * 16 + l16) * HDc + c * 32, lane);

    const v16bf ones = ones_frag();

    float m[8], l[8];
    v8f o[8];
#pragma unroll
    for (int r = 0; r < 8; ++r) { m[r] = -1e30f; l[r] = 0.f; }
#pragma unroll
    for (int f = 0; f < 8; ++f) o[f] = vzero8();

    const int nfull = (qb * 16 + 1) / 32;        // fully-unmasked key blocks
    const int nkb   = (qb * 16 + 15) / 32 + 1;   // total key blocks (causal)
    for (int j = 0; j < nfull; ++j)
        attn_kblock<false>(j * 32, qb, lane, l16, half, kOff,
                           kBase, vBase, qf, ones, o, m, l, pLds);
    for (int j = nfull; j < nkb; ++j)
        attn_kblock<true>(j * 32, qb, lane, l16, half, kOff,
                          kBase, vBase, qf, ones, o, m, l, pLds);

    // epilogue: normalize and store bf16 into [B*S, H*HD]
#pragma unroll
    for (int f = 0; f < 8; ++f) {
#pragma unroll
        for (int r = 0; r < 8; ++r) {
            const int row = qb * 16 + half + r;
            float ov = o[f][r] / l[r];
            ao[((size_t)b * Sc + row) * (size_t)DIMc + h * HDc + f * 16 + l16] =
                (bf16_t)ov;
        }
    }
}

// ---------------------------------------------------------------------------
// Host-side orchestration
// ---------------------------------------------------------------------------
extern "C" void kernel_launch(void* const* d_in, const int* in_sizes, int n_in,
                              void* d_out, int out_size, void* d_ws, size_t ws_size,
                              hipStream_t stream) {
    (void)in_sizes; (void)n_in; (void)out_size; (void)ws_size;
    const float* x    = (const float*)d_in[0];
    const float* cosp = (const float*)d_in[1];
    const float* sinp = (const float*)d_in[2];
    const float* wq   = (const float*)d_in[3];
    const float* wk   = (const float*)d_in[4];
    const float* wv   = (const float*)d_in[5];
    const float* wo   = (const float*)d_in[6];
    float* out = (float*)d_out;

    char* ws = (char*)d_ws;
    size_t off = 0;
    auto carve = [&](size_t elems) -> bf16_t* {
        bf16_t* p = (bf16_t*)(ws + off);
        off += ((elems * sizeof(bf16_t)) + 255) & ~(size_t)255;
        return p;
    };

    const size_t nX  = (size_t)Mrows * DIMc;            // 8.4M
    const size_t nKV = (size_t)Mrows * NKVc * HDc;      // 2.1M
    bf16_t* xb  = carve(nX);
    bf16_t* wqT = carve((size_t)DIMc * DIMc);
    bf16_t* wkT = carve((size_t)(NKVc * HDc) * DIMc);
    bf16_t* wvT = carve((size_t)(NKVc * HDc) * DIMc);
    bf16_t* woT = carve((size_t)DIMc * DIMc);
    bf16_t* q0  = carve(nX);
    bf16_t* k0  = carve(nKV);
    bf16_t* v0  = carve(nKV);
    bf16_t* qr  = carve(nX);
    bf16_t* kr  = carve(nKV);
    bf16_t* vt  = carve(nKV);
    bf16_t* ao  = carve(nX);

    auto blocks = [](size_t n) { return (unsigned)((n + 255) / 256); };

    // 1) casts + weight transposes
    cast_f32_bf16<<<blocks(nX), 256, 0, stream>>>(x, xb, nX);
    cast_transpose<<<blocks((size_t)DIMc * DIMc), 256, 0, stream>>>(wq, wqT, DIMc, DIMc);
    cast_transpose<<<blocks((size_t)DIMc * NKVc * HDc), 256, 0, stream>>>(wk, wkT, DIMc, NKVc * HDc);
    cast_transpose<<<blocks((size_t)DIMc * NKVc * HDc), 256, 0, stream>>>(wv, wvT, DIMc, NKVc * HDc);
    cast_transpose<<<blocks((size_t)DIMc * DIMc), 256, 0, stream>>>(wo, woT, DIMc, DIMc);

    // 2) QKV projections (bf16 WMMA, f32 accumulate); workgroup tile 64x256
    gemm_bf16_k<bf16_t><<<dim3(DIMc / 256, Mrows / 64), 256, 0, stream>>>(
        xb, wqT, q0, Mrows, DIMc, DIMc);
    gemm_bf16_k<bf16_t><<<dim3((NKVc * HDc) / 256, Mrows / 64), 256, 0, stream>>>(
        xb, wkT, k0, Mrows, NKVc * HDc, DIMc);
    gemm_bf16_k<bf16_t><<<dim3((NKVc * HDc) / 256, Mrows / 64), 256, 0, stream>>>(
        xb, wvT, v0, Mrows, NKVc * HDc, DIMc);

    // 3) RoPE + layout rearrangement (1/sqrt(HD) folded into Q here)
    rope_rearrange<<<blocks((size_t)Bc * Sc * NHc * (HDc / 2)), 256, 0, stream>>>(
        q0, cosp, sinp, qr, NHc, SCALEc);
    rope_rearrange<<<blocks((size_t)Bc * Sc * NKVc * (HDc / 2)), 256, 0, stream>>>(
        k0, cosp, sinp, kr, NKVc, 1.0f);
    v_transpose<<<blocks((size_t)Bc * Sc * NKVc * HDc), 256, 0, stream>>>(v0, vt);

    // 4) causal flash attention (WMMA for QK^T and PV)
    attn_fwd<<<dim3(Sc / 16, NHc, Bc), 32, 0, stream>>>(qr, kr, vt, ao);

    // 5) output projection -> fp32
    gemm_bf16_k<float><<<dim3(DIMc / 256, Mrows / 64), 256, 0, stream>>>(
        ao, woT, out, Mrows, DIMc, DIMc);
}
